// GRU_Classifier_69114613730636
// MI455X (gfx1250) — compile-verified, matrix operand order checked by reading, not run on hardware
//
#include <hip/hip_runtime.h>
#include <hip/hip_bf16.h>

typedef __bf16 bf16_t;
typedef __attribute__((ext_vector_type(16))) __bf16 v16bf;
typedef __attribute__((ext_vector_type(8)))  __bf16 v8bf;
typedef __attribute__((ext_vector_type(8)))  float  v8f;

#define B_SZ   2048
#define IN_SZ  512
#define H_SZ   512
#define OUT_SZ 256
#define T_MAX  64

// ---------------------------------------------------------------- helpers ---

__device__ __forceinline__ bf16_t f32_to_bf16(float f) {
  unsigned u = __float_as_uint(f);
  u += 0x7FFFu + ((u >> 16) & 1u);            // round-to-nearest-even
  unsigned short hs = (unsigned short)(u >> 16);
  union { unsigned short s; bf16_t b; } cv;
  cv.s = hs;
  return cv.b;
}

// A-frag / B-frag loader for v_wmma_f32_16x16x32_bf16.
// Lane 0-15 indexes M (or N for B); laneHi selects which K-halves this lane
// owns: elements 0..7 -> K = kb + laneHi*8 .. +7, elements 8..15 -> K = kb+16+laneHi*8 .. +7
__device__ __forceinline__ v16bf load_frag(const bf16_t* __restrict__ row, int kb, int laneHi) {
  v8bf lo = *(const v8bf*)(row + kb + laneHi * 8);        // 16B load
  v8bf hi = *(const v8bf*)(row + kb + 16 + laneHi * 8);   // 16B load
  v16bf r;
#pragma unroll
  for (int i = 0; i < 8; ++i) { r[i] = lo[i]; r[i + 8] = hi[i]; }
  return r;
}

#define WMMA_BF16(a, b, c) \
  __builtin_amdgcn_wmma_f32_16x16x32_bf16(false, (a), false, (b), (short)0, (c), false, false)

// ---------------------------------------------------------------- kernels ---

// C[M,N] = A[M,K] @ W[N,K]^T   (bf16 inputs, f32 out)
// grid = (N/64, M/128), block = 128 (4 waves); each wave computes a 32x64 tile:
// 2 A-frags + 4 W-frags (12 b128 loads) feed 8 WMMAs per K-step.
__global__ __launch_bounds__(128)
void gemm_bf16(const bf16_t* __restrict__ A, const bf16_t* __restrict__ W,
               float* __restrict__ C, int N, int K,
               const int* __restrict__ Tdev, int t)
{
  if (t >= *Tdev) return;                       // wave-uniform: EXEC stays all-1s
  const int lane   = threadIdx.x & 31;
  const int wave   = threadIdx.x >> 5;
  const int laneLo = lane & 15;
  const int laneHi = lane >> 4;
  const int mBase  = blockIdx.y * 128 + wave * 32;
  const int nBase  = blockIdx.x * 64;

  const bf16_t* arow0 = A + (size_t)(mBase +  0 + laneLo) * K;
  const bf16_t* arow1 = A + (size_t)(mBase + 16 + laneLo) * K;
  const bf16_t* w0    = W + (size_t)(nBase +  0 + laneLo) * K;
  const bf16_t* w1    = W + (size_t)(nBase + 16 + laneLo) * K;
  const bf16_t* w2    = W + (size_t)(nBase + 32 + laneLo) * K;
  const bf16_t* w3    = W + (size_t)(nBase + 48 + laneLo) * K;

  v8f c00 = {}, c01 = {}, c02 = {}, c03 = {};
  v8f c10 = {}, c11 = {}, c12 = {}, c13 = {};
  for (int kb = 0; kb < K; kb += 32) {
    v16bf a0 = load_frag(arow0, kb, laneHi);
    v16bf a1 = load_frag(arow1, kb, laneHi);
    v16bf b0 = load_frag(w0,    kb, laneHi);
    v16bf b1 = load_frag(w1,    kb, laneHi);
    v16bf b2 = load_frag(w2,    kb, laneHi);
    v16bf b3 = load_frag(w3,    kb, laneHi);
    c00 = WMMA_BF16(a0, b0, c00);
    c01 = WMMA_BF16(a0, b1, c01);
    c02 = WMMA_BF16(a0, b2, c02);
    c03 = WMMA_BF16(a0, b3, c03);
    c10 = WMMA_BF16(a1, b0, c10);
    c11 = WMMA_BF16(a1, b1, c11);
    c12 = WMMA_BF16(a1, b2, c12);
    c13 = WMMA_BF16(a1, b3, c13);
  }
#pragma unroll
  for (int r = 0; r < 8; ++r) {
    size_t m0 = (size_t)(mBase + r + laneHi * 8);
    float* crow0 = C + m0 * N + nBase + laneLo;
    crow0[0]  = c00[r];
    crow0[16] = c01[r];
    crow0[32] = c02[r];
    crow0[48] = c03[r];
    size_t m1 = m0 + 16;
    float* crow1 = C + m1 * N + nBase + laneLo;
    crow1[0]  = c10[r];
    crow1[16] = c11[r];
    crow1[32] = c12[r];
    crow1[48] = c13[r];
  }
}

// logits = h1 @ Wout^T + bout, scattered: batch row i written to out row
// starts[i]+t iff counts[i] > t.  grid = (OUT/64=4, B/128=16), block = 128.
__global__ __launch_bounds__(128)
void out_gemm_bf16(const bf16_t* __restrict__ A, const bf16_t* __restrict__ W,
                   const float* __restrict__ bout,
                   const int* __restrict__ counts, const int* __restrict__ starts,
                   float* __restrict__ out, int K,
                   const int* __restrict__ Tdev, int t)
{
  if (t >= *Tdev) return;
  const int lane   = threadIdx.x & 31;
  const int wave   = threadIdx.x >> 5;
  const int laneLo = lane & 15;
  const int laneHi = lane >> 4;
  const int mBase  = blockIdx.y * 128 + wave * 32;
  const int nBase  = blockIdx.x * 64;

  const bf16_t* arow0 = A + (size_t)(mBase +  0 + laneLo) * K;
  const bf16_t* arow1 = A + (size_t)(mBase + 16 + laneLo) * K;
  const bf16_t* w0    = W + (size_t)(nBase +  0 + laneLo) * K;
  const bf16_t* w1    = W + (size_t)(nBase + 16 + laneLo) * K;
  const bf16_t* w2    = W + (size_t)(nBase + 32 + laneLo) * K;
  const bf16_t* w3    = W + (size_t)(nBase + 48 + laneLo) * K;

  v8f c00 = {}, c01 = {}, c02 = {}, c03 = {};
  v8f c10 = {}, c11 = {}, c12 = {}, c13 = {};
  for (int kb = 0; kb < K; kb += 32) {
    v16bf a0 = load_frag(arow0, kb, laneHi);
    v16bf a1 = load_frag(arow1, kb, laneHi);
    v16bf b0 = load_frag(w0,    kb, laneHi);
    v16bf b1 = load_frag(w1,    kb, laneHi);
    v16bf b2 = load_frag(w2,    kb, laneHi);
    v16bf b3 = load_frag(w3,    kb, laneHi);
    c00 = WMMA_BF16(a0, b0, c00);
    c01 = WMMA_BF16(a0, b1, c01);
    c02 = WMMA_BF16(a0, b2, c02);
    c03 = WMMA_BF16(a0, b3, c03);
    c10 = WMMA_BF16(a1, b0, c10);
    c11 = WMMA_BF16(a1, b1, c11);
    c12 = WMMA_BF16(a1, b2, c12);
    c13 = WMMA_BF16(a1, b3, c13);
  }
  const int n = nBase + laneLo;
  const float bo0 = bout[n], bo1 = bout[n + 16], bo2 = bout[n + 32], bo3 = bout[n + 48];
#pragma unroll
  for (int r = 0; r < 8; ++r) {
    int gm0 = mBase + r + laneHi * 8;
    if (counts[gm0] > t) {                               // this (batch,t) is selected
      float* orow = out + (size_t)(starts[gm0] + t) * OUT_SZ + n;
      orow[0]  = c00[r] + bo0;
      orow[16] = c01[r] + bo1;
      orow[32] = c02[r] + bo2;
      orow[48] = c03[r] + bo3;
    }
    int gm1 = gm0 + 16;
    if (counts[gm1] > t) {
      float* orow = out + (size_t)(starts[gm1] + t) * OUT_SZ + n;
      orow[0]  = c10[r] + bo0;
      orow[16] = c11[r] + bo1;
      orow[32] = c12[r] + bo2;
      orow[48] = c13[r] + bo3;
    }
  }
}

// GRU gate fusion: reads Xp = x@Wih^T (no bias) and Hh = h@Whh^T (no bias),
// applies biases + sigmoid/tanh, updates h (f32) and h_bf (bf16 for next GEMM).
// grid = B, block = H
__global__ __launch_bounds__(512)
void gru_gates(const float* __restrict__ Xp, const float* __restrict__ Hh,
               const float* __restrict__ bih, const float* __restrict__ bhh,
               float* __restrict__ h, bf16_t* __restrict__ h_bf,
               const int* __restrict__ Tdev, int t)
{
  if (t >= *Tdev) return;
  const int i = blockIdx.x;
  const int j = threadIdx.x;
  const int H = H_SZ;
  const size_t b3 = (size_t)i * 3 * H;
  float xr = Xp[b3 + j]         + bih[j];
  float xz = Xp[b3 + H + j]     + bih[H + j];
  float xn = Xp[b3 + 2 * H + j] + bih[2 * H + j];
  float hr = Hh[b3 + j]         + bhh[j];
  float hz = Hh[b3 + H + j]     + bhh[H + j];
  float hn = Hh[b3 + 2 * H + j] + bhh[2 * H + j];
  float rg = 1.0f / (1.0f + __expf(-(xr + hr)));
  float zg = 1.0f / (1.0f + __expf(-(xz + hz)));
  float ng = tanhf(xn + rg * hn);
  size_t idx = (size_t)i * H + j;
  float hprev = h[idx];
  float hnew = (1.0f - zg) * ng + zg * hprev;
  h[idx] = hnew;
  h_bf[idx] = f32_to_bf16(hnew);
}

// ------------------------------------------------------------ setup kernels -

__global__ void k_zero_i32(int* p, int n) {
  int i = blockIdx.x * blockDim.x + threadIdx.x;
  if (i < n) p[i] = 0;
}
__global__ void k_zero_f32(float* p, int n) {
  int i = blockIdx.x * blockDim.x + threadIdx.x;
  if (i < n) p[i] = 0.0f;
}
__global__ void k_zero_bf16(bf16_t* p, int n) {
  int i = blockIdx.x * blockDim.x + threadIdx.x;
  if (i < n) ((unsigned short*)p)[i] = 0;
}
__global__ void k_cvt_bf16(const float* __restrict__ s, bf16_t* __restrict__ d, int n) {
  int i = blockIdx.x * blockDim.x + threadIdx.x;
  if (i < n) d[i] = f32_to_bf16(s[i]);
}
__global__ void k_count(const int* __restrict__ idx, int* __restrict__ counts, int n) {
  int i = blockIdx.x * blockDim.x + threadIdx.x;
  if (i < n) atomicAdd(&counts[idx[i]], 1);
}
// serial exclusive scan + max (B=2048, trivial cost, runs once)
__global__ void k_scan(const int* __restrict__ counts, int* __restrict__ starts,
                       int* __restrict__ Tdev, int nb) {
  if (threadIdx.x == 0 && blockIdx.x == 0) {
    int acc = 0, mx = 0;
    for (int i = 0; i < nb; ++i) {
      starts[i] = acc;
      int c = counts[i];
      acc += c;
      if (c > mx) mx = c;
    }
    *Tdev = mx;
  }
}

// ------------------------------------------------------------------ launch --

static inline char* carve(char*& p, size_t bytes) {
  char* r = p;
  p += (bytes + 255) & ~(size_t)255;
  return r;
}

extern "C" void kernel_launch(void* const* d_in, const int* in_sizes, int n_in,
                              void* d_out, int out_size, void* d_ws, size_t ws_size,
                              hipStream_t stream) {
  const float* x    = (const float*)d_in[0];
  const float* Wih0 = (const float*)d_in[1];
  const float* Whh0 = (const float*)d_in[2];
  const float* bih0 = (const float*)d_in[3];
  const float* bhh0 = (const float*)d_in[4];
  const float* Wih1 = (const float*)d_in[5];
  const float* Whh1 = (const float*)d_in[6];
  const float* bih1 = (const float*)d_in[7];
  const float* bhh1 = (const float*)d_in[8];
  const float* Wout = (const float*)d_in[9];
  const float* bout = (const float*)d_in[10];
  const int*   psi  = (const int*)d_in[11];
  const int    Nidx = in_sizes[11];
  float* out = (float*)d_out;

  const int B = B_SZ, H = H_SZ, IN = IN_SZ;
  const int G = 3 * H;                       // 1536

  char* p = (char*)d_ws;
  int*    counts  = (int*)carve(p, (size_t)B * 4);
  int*    starts  = (int*)carve(p, (size_t)B * 4);
  int*    Tdev    = (int*)carve(p, 4);
  bf16_t* x_bf    = (bf16_t*)carve(p, (size_t)B * IN * 2);
  bf16_t* Wih0_bf = (bf16_t*)carve(p, (size_t)G * IN * 2);
  bf16_t* Whh0_bf = (bf16_t*)carve(p, (size_t)G * H * 2);
  bf16_t* Wih1_bf = (bf16_t*)carve(p, (size_t)G * H * 2);
  bf16_t* Whh1_bf = (bf16_t*)carve(p, (size_t)G * H * 2);
  bf16_t* Wout_bf = (bf16_t*)carve(p, (size_t)OUT_SZ * H * 2);
  float*  Xp0     = (float*)carve(p, (size_t)B * G * 4);   // x@Wih0^T, reused every step
  float*  bufA    = (float*)carve(p, (size_t)B * G * 4);   // Hh0, then Xp1
  float*  bufB    = (float*)carve(p, (size_t)B * G * 4);   // Hh1
  float*  h0      = (float*)carve(p, (size_t)B * H * 4);
  float*  h1      = (float*)carve(p, (size_t)B * H * 4);
  bf16_t* h0_bf   = (bf16_t*)carve(p, (size_t)B * H * 2);
  bf16_t* h1_bf   = (bf16_t*)carve(p, (size_t)B * H * 2);

  // --- setup: histogram, scan, weight/input conversion, zero state ---
  k_zero_i32<<<(B + 255) / 256, 256, 0, stream>>>(counts, B);
  k_count<<<(Nidx + 255) / 256, 256, 0, stream>>>(psi, counts, Nidx);
  k_scan<<<1, 1, 0, stream>>>(counts, starts, Tdev, B);

  k_cvt_bf16<<<(B * IN + 255) / 256, 256, 0, stream>>>(x, x_bf, B * IN);
  k_cvt_bf16<<<(G * IN + 255) / 256, 256, 0, stream>>>(Wih0, Wih0_bf, G * IN);
  k_cvt_bf16<<<(G * H + 255) / 256, 256, 0, stream>>>(Whh0, Whh0_bf, G * H);
  k_cvt_bf16<<<(G * H + 255) / 256, 256, 0, stream>>>(Wih1, Wih1_bf, G * H);
  k_cvt_bf16<<<(G * H + 255) / 256, 256, 0, stream>>>(Whh1, Whh1_bf, G * H);
  k_cvt_bf16<<<(OUT_SZ * H + 255) / 256, 256, 0, stream>>>(Wout, Wout_bf, OUT_SZ * H);

  k_zero_f32<<<(B * H + 255) / 256, 256, 0, stream>>>(h0, B * H);
  k_zero_f32<<<(B * H + 255) / 256, 256, 0, stream>>>(h1, B * H);
  k_zero_bf16<<<(B * H + 255) / 256, 256, 0, stream>>>(h0_bf, B * H);
  k_zero_bf16<<<(B * H + 255) / 256, 256, 0, stream>>>(h1_bf, B * H);

  dim3 gemm_grid(G / 64, B / 128);      // (24, 16)
  dim3 out_grid(OUT_SZ / 64, B / 128);  // (4, 16)

  // Layer-0 input projection is timestep-invariant: compute once.
  gemm_bf16<<<gemm_grid, 128, 0, stream>>>(x_bf, Wih0_bf, Xp0, G, IN, Tdev, 0);

  // --- time loop (T is dynamic, read on device; extra steps early-exit) ---
  for (int t = 0; t < T_MAX; ++t) {
    // layer 0: Hh0 = h0 @ Whh0^T ; gates
    gemm_bf16<<<gemm_grid, 128, 0, stream>>>(h0_bf, Whh0_bf, bufA, G, H, Tdev, t);
    gru_gates<<<B, H, 0, stream>>>(Xp0, bufA, bih0, bhh0, h0, h0_bf, Tdev, t);
    // layer 1: Xp1 = y0 @ Wih1^T ; Hh1 = h1 @ Whh1^T ; gates
    gemm_bf16<<<gemm_grid, 128, 0, stream>>>(h0_bf, Wih1_bf, bufA, G, H, Tdev, t);
    gemm_bf16<<<gemm_grid, 128, 0, stream>>>(h1_bf, Whh1_bf, bufB, G, H, Tdev, t);
    gru_gates<<<B, H, 0, stream>>>(bufA, bufB, bih1, bhh1, h1, h1_bf, Tdev, t);
    // fused gather + output projection: scatter rows with counts[i] > t
    out_gemm_bf16<<<out_grid, 128, 0, stream>>>(h1_bf, Wout_bf, bout, counts, starts,
                                                out, H, Tdev, t);
  }
}
